// JointRecog_2585570312222
// MI455X (gfx1250) — matrix operand order repaired and checked by reading, not compiled, not run on hardware
//
#include <hip/hip_runtime.h>
#include <hip/hip_bf16.h>
#include <math.h>

typedef __attribute__((ext_vector_type(16))) _Float16 v16h;
typedef __attribute__((ext_vector_type(8)))  float    v8f;

#define NTOT 65536
#define SEQ  2048
#define PED  32
#define TT   8
#define HID  64
#define NDIM 8

// ---------------- WMMA helpers (CDNA5 gfx1250, wave32) ----------------

static __device__ __forceinline__ v8f wmma16(v16h a, v16h b, v8f c) {
  // D = A(16x32 f16) * B(32x16 f16) + C(16x16 f32)
  return __builtin_amdgcn_wmma_f32_16x16x32_f16(false, a, false, b, (short)0, c, false, false);
}

// Packed-B global load: 16 contiguous halves (32B) per lane.
static __device__ __forceinline__ v16h load_pk(const _Float16* pk, int tci, int lane) {
  return *(const v16h*)(pk + ((size_t)tci * 32 + lane) * 16);
}

// Gather a 16x32 f16 A operand from a row-major [16][stride] half buffer in LDS.
// ISA A-layout: lane L row = L%16; element e -> K = (e&1)+2*((e>>1)&3)+8*(L>>4)+16*(e>>3).
static __device__ __forceinline__ v16h ldsA(const _Float16* buf, int stride, int lane, int kOff) {
  int r = lane & 15, hh = lane >> 4;
  const _Float16* row = buf + r * stride;
  v16h a;
#pragma unroll
  for (int p = 0; p < 8; ++p) {
    int k = kOff + 2 * (p & 3) + 8 * hh + 16 * (p >> 2);
    a[2 * p]     = row[k];
    a[2 * p + 1] = row[k + 1];
  }
  return a;
}

static __device__ __forceinline__ float lane_xor(float v, int mask) {
  int lane = (int)(threadIdx.x & 31);
  int i = __builtin_amdgcn_ds_bpermute((lane ^ mask) << 2, __builtin_bit_cast(int, v));
  return __builtin_bit_cast(float, i);
}

static __device__ __forceinline__ float sigm(float x) { return 1.0f / (1.0f + expf(-x)); }

// ---------------- weight pack kernel: W[out][in] -> B-layout tiles ----------------
// dst[(tile*nChunks+chunk)*32 + lane][e] = W[tile*16 + lane%16][chunk*32 + 8*(lane/16) + (e&7) + 16*(e>>3)]
__global__ __launch_bounds__(32) void k_packB(const float* __restrict__ W, _Float16* __restrict__ dst,
                                              int outDim, int inDim, int nChunks) {
  int tile  = blockIdx.x;
  int chunk = blockIdx.y;
  int lane  = threadIdx.x;
  int col   = tile * 16 + (lane & 15);
  int kbase = chunk * 32 + 8 * (lane >> 4);
  v16h v;
#pragma unroll
  for (int e = 0; e < 16; ++e) {
    int k = kbase + (e & 7) + 16 * (e >> 3);
    float f = (col < outDim && k < inDim) ? W[(size_t)col * inDim + k] : 0.0f;
    v[e] = (_Float16)f;
  }
  *(v16h*)(dst + ((size_t)(tile * nChunks + chunk) * 32 + lane) * 16) = v;
}

// ---------------- LSTM encoder: emb + 8 steps, 16 rows per wave ----------------
__global__ __launch_bounds__(256) void k_lstm(const float* __restrict__ x_y_full,
                                              const float* __restrict__ W_emb,
                                              const float* __restrict__ b_emb,
                                              const float* __restrict__ b_ih,
                                              const float* __restrict__ b_hh,
                                              const _Float16* __restrict__ pkWih,
                                              const _Float16* __restrict__ pkWhh,
                                              float* __restrict__ xe,
                                              _Float16* __restrict__ xepk) {
  __shared__ float    s_we[128];     // W_emb [64][2]
  __shared__ float    s_be[64];
  __shared__ float    s_bias[256];   // b_ih + b_hh
  __shared__ _Float16 s_h[8][16 * 64];

  int tid = threadIdx.x;
  for (int i = tid; i < 128; i += 256) s_we[i] = W_emb[i];
  for (int i = tid; i < 64;  i += 256) s_be[i] = b_emb[i];
  for (int i = tid; i < 256; i += 256) s_bias[i] = b_ih[i] + b_hh[i];
  __syncthreads();

  int wave = tid >> 5, lane = tid & 31;
  int tile = blockIdx.x * 8 + wave;       // 4096 tiles of 16 rows
  int row0 = tile * 16;
  int r = lane & 15, hh = lane >> 4;
  _Float16* hbuf = &s_h[wave][0];

  const float* xrow = x_y_full + (size_t)(row0 + r) * TT * 2;

  float cstate[32];
#pragma unroll
  for (int i = 0; i < 32; ++i) cstate[i] = 0.0f;

#pragma unroll 1
  for (int t = 0; t < TT; ++t) {
    float x0 = xrow[t * 2 + 0], x1 = xrow[t * 2 + 1];
    // build A_emb (two 16x32 chunks) directly in A-layout
    v16h aemb0, aemb1;
#pragma unroll
    for (int e = 0; e < 16; ++e) {
      int kk = (e & 1) + 2 * ((e >> 1) & 3) + 8 * hh + 16 * (e >> 3);
      aemb0[e] = (_Float16)(x0 * s_we[2 * kk] + x1 * s_we[2 * kk + 1] + s_be[kk]);
      int k2 = kk + 32;
      aemb1[e] = (_Float16)(x0 * s_we[2 * k2] + x1 * s_we[2 * k2 + 1] + s_be[k2]);
    }
    v16h ah0 = {}, ah1 = {};
    if (t > 0) { ah0 = ldsA(hbuf, 64, lane, 0); ah1 = ldsA(hbuf, 64, lane, 32); }

    // keep this loop rolled: 4 gate accumulators + 4 A operands stay < 256 VGPRs
#pragma unroll 1
    for (int q = 0; q < 4; ++q) {
      v8f gacc[4];
#pragma unroll
      for (int g = 0; g < 4; ++g) {
        float bv = s_bias[g * 64 + q * 16 + r];
        v8f acc = {bv, bv, bv, bv, bv, bv, bv, bv};
        int ti = (g * 4 + q) * 2;
        acc = wmma16(aemb0, load_pk(pkWih, ti + 0, lane), acc);
        acc = wmma16(aemb1, load_pk(pkWih, ti + 1, lane), acc);
        if (t > 0) {
          acc = wmma16(ah0, load_pk(pkWhh, ti + 0, lane), acc);
          acc = wmma16(ah1, load_pk(pkWhh, ti + 1, lane), acc);
        }
        gacc[g] = acc;
      }
      int hu = q * 16 + r;
#pragma unroll
      for (int v = 0; v < 8; ++v) {
        float iv = sigm(gacc[0][v]);
        float fv = sigm(gacc[1][v]);
        float gv = tanhf(gacc[2][v]);
        float ov = sigm(gacc[3][v]);
        float cc = fv * cstate[q * 8 + v] + iv * gv;
        cstate[q * 8 + v] = cc;
        float hv = ov * tanhf(cc);
        int m = v + 8 * hh;
        hbuf[m * 64 + hu] = (_Float16)hv;
        if (t == TT - 1) xe[(size_t)(row0 + m) * 64 + hu] = hv;
      }
    }
  }
  // export A-packed f16 xe for later GEMMs
  v16h p0 = ldsA(hbuf, 64, lane, 0);
  v16h p1 = ldsA(hbuf, 64, lane, 32);
  *(v16h*)(xepk + ((size_t)(tile * 2 + 0) * 32 + lane) * 16) = p0;
  *(v16h*)(xepk + ((size_t)(tile * 2 + 1) * 32 + lane) * 16) = p1;
}

// ---------------- ae / ac / ze projections (64->64 each) ----------------
__global__ __launch_bounds__(256) void k_proj(const _Float16* __restrict__ xepk,
                                              const _Float16* __restrict__ pk0, const float* __restrict__ b0, float* __restrict__ o0,
                                              const _Float16* __restrict__ pk1, const float* __restrict__ b1_, float* __restrict__ o1,
                                              const _Float16* __restrict__ pk2, const float* __restrict__ b2_, float* __restrict__ o2) {
  int tid = threadIdx.x, wave = tid >> 5, lane = tid & 31;
  int tile = blockIdx.x * 8 + wave;
  int row0 = tile * 16;
  int r = lane & 15, hh = lane >> 4;
  v16h a0 = *(const v16h*)(xepk + ((size_t)(tile * 2 + 0) * 32 + lane) * 16);
  v16h a1 = *(const v16h*)(xepk + ((size_t)(tile * 2 + 1) * 32 + lane) * 16);
  const _Float16* pks[3] = {pk0, pk1, pk2};
  const float*    bs[3]  = {b0, b1_, b2_};
  float*          os[3]  = {o0, o1, o2};
#pragma unroll 1
  for (int mi = 0; mi < 3; ++mi) {
#pragma unroll 1
    for (int ct = 0; ct < 4; ++ct) {
      float bv = bs[mi][ct * 16 + r];
      v8f acc = {bv, bv, bv, bv, bv, bv, bv, bv};
      acc = wmma16(a0, load_pk(pks[mi], ct * 2 + 0, lane), acc);
      acc = wmma16(a1, load_pk(pks[mi], ct * 2 + 1, lane), acc);
#pragma unroll
      for (int v = 0; v < 8; ++v)
        os[mi][(size_t)(row0 + v + 8 * hh) * 64 + ct * 16 + r] = acc[v];
    }
  }
}

// ---------------- per-sequence social attention: scores -> softmax -> att ----------------
__global__ __launch_bounds__(256) void k_att(const float* __restrict__ xe,
                                             const float* __restrict__ xl,
                                             const float* __restrict__ ae,
                                             const float* __restrict__ ac,
                                             const float* __restrict__ Wxy_l,
                                             const float* __restrict__ bxy_l,
                                             const float* __restrict__ Wxy_f,
                                             const float* __restrict__ bxy_f,
                                             float* __restrict__ att) {
  int s = blockIdx.x;
  __shared__ float sxe[32][64];
  __shared__ float sae[32][64];
  __shared__ float sac[32][64];
  __shared__ float sxl[32][2];
  __shared__ float swl[128];
  __shared__ float sbl[64];
  __shared__ float swf[64];
  __shared__ float smask[32];
  __shared__ float ssc[32][32];
  __shared__ float sal[32][32];
  int tid = threadIdx.x;
  for (int i = tid; i < 2048; i += 256) {
    sxe[i >> 6][i & 63] = xe[(size_t)s * 2048 + i];
    sae[i >> 6][i & 63] = ae[(size_t)s * 2048 + i];
    sac[i >> 6][i & 63] = ac[(size_t)s * 2048 + i];
  }
  for (int i = tid; i < 64;  i += 256) { sxl[i >> 1][i & 1] = xl[(size_t)s * 64 + i]; sbl[i] = bxy_l[i]; swf[i] = Wxy_f[i]; }
  for (int i = tid; i < 128; i += 256) swl[i] = Wxy_l[i];
  __syncthreads();
  if (tid < 32) {
    float sum = 0.0f;
    for (int d = 0; d < 64; ++d) sum += fabsf(sxe[tid][d]);
    smask[tid] = (sum > 0.0f) ? 1.0f : 0.0f;
  }
  __syncthreads();
  float bf = bxy_f[0];
  for (int idx = tid; idx < 1024; idx += 256) {
    int i = idx >> 5, jj = idx & 31;
    float rx = sxl[jj][0] - sxl[i][0];
    float ry = sxl[jj][1] - sxl[i][1];
    float acc = bf;
    for (int d = 0; d < 64; ++d)
      acc += swf[d] * tanhf(sae[jj][d] + sac[i][d] + swl[2 * d] * rx + swl[2 * d + 1] * ry + sbl[d]);
    ssc[i][jj] = acc;
  }
  __syncthreads();
  if (tid < 32) {
    int i = tid;
    float mx = -1e38f;
    for (int jj = 0; jj < 32; ++jj) if (smask[jj] > 0.0f) mx = fmaxf(mx, ssc[i][jj]);
    float den = 0.0f;
    for (int jj = 0; jj < 32; ++jj) {
      float e = (smask[jj] > 0.0f) ? expf(ssc[i][jj] - mx) : 0.0f;
      sal[i][jj] = e;
      den += e;
    }
    float inv = (den > 0.0f) ? 1.0f / den : 0.0f;
    for (int jj = 0; jj < 32; ++jj) sal[i][jj] *= inv;
  }
  __syncthreads();
  for (int idx = tid; idx < 2048; idx += 256) {
    int i = idx >> 6, d = idx & 63;
    float acc = 0.0f;
    for (int jj = 0; jj < 32; ++jj) acc += sal[i][jj] * sxe[jj][d];
    att[((size_t)s * 32 + i) * 64 + d] = acc;
  }
}

// ---------------- decoder step j: causal z-attention + MLP (WMMA) ----------------
__global__ __launch_bounds__(128) void k_step(int j,
                                              const float* __restrict__ xe,
                                              const float* __restrict__ xl,
                                              const float* __restrict__ ze,
                                              const float* __restrict__ att,
                                              const float* __restrict__ zsbuf,
                                              const float* __restrict__ Wz_l,
                                              const float* __restrict__ bz_l,
                                              const float* __restrict__ Wz_f,
                                              const float* __restrict__ bz_f,
                                              const _Float16* __restrict__ pkW1,
                                              const float* __restrict__ b1,
                                              const _Float16* __restrict__ pkW2,
                                              const float* __restrict__ b2,
                                              const float* __restrict__ noise,
                                              float* __restrict__ outMeans,
                                              float* __restrict__ outLogs,
                                              float* __restrict__ outZs,
                                              float* __restrict__ zswr) {
  __shared__ _Float16 arow[4][16][224];   // per-wave feature rows, K padded to 224
  __shared__ _Float16 hbuf[4][16][128];   // per-wave hidden transpose
  __shared__ float    ssc[4][16][32];     // per-wave scores/weights

  int tid = threadIdx.x, wave = tid >> 5, lane = tid & 31;
  int s0 = blockIdx.x * 64 + wave * 16;   // sequence rows handled by this wave
  int r = lane & 15, hh = lane >> 4;

  // fill xe / att parts of the feature rows
  for (int idx = lane; idx < 16 * 128; idx += 32) {
    int row = idx >> 7, d = idx & 127;
    size_t n = (size_t)(s0 + row) * 32 + j;
    float v = (d < 64) ? xe[n * 64 + d] : att[n * 64 + (d - 64)];
    arow[wave][row][d] = (_Float16)v;
  }
  for (int idx = lane; idx < 16 * 96; idx += 32) {
    int row = idx / 96, d = idx % 96;
    arow[wave][row][128 + d] = (_Float16)0.0f;   // z_prev default + K padding
  }

  if (j > 0) {
    int srow = s0 + r;                    // this lane's sequence; hh selects dim-half
    float xlj0 = xl[((size_t)srow * 32 + j) * 2 + 0];
    float xlj1 = xl[((size_t)srow * 32 + j) * 2 + 1];
    float bzf = bz_f[0];
#pragma unroll 1
    for (int k = 0; k < j; ++k) {
      if (k + 1 < j) {   // prefetch next key's rows (global_prefetch_b8)
        __builtin_prefetch(ze + ((size_t)srow * 32 + k + 1) * 64 + hh * 32, 0, 3);
        __builtin_prefetch(xe + ((size_t)srow * 32 + k + 1) * 64 + hh * 36, 0, 3);
      }
      float rx = xl[((size_t)srow * 32 + k) * 2 + 0] - xlj0;
      float ry = xl[((size_t)srow * 32 + k) * 2 + 1] - xlj1;
      const float* zek = ze + ((size_t)srow * 32 + k) * 64;
      float part = 0.0f;
      for (int d = hh * 32; d < hh * 32 + 32; ++d)
        part += Wz_f[d] * tanhf(zek[d] + Wz_l[2 * d] * rx + Wz_l[2 * d + 1] * ry + bz_l[d]);
      float mp = 0.0f;
      const float* xek = xe + ((size_t)srow * 32 + k) * 64;
      const float* zsk = zsbuf + ((size_t)srow * 32 + k) * 8;
      for (int dd = 0; dd < 36; ++dd) {
        int d = hh * 36 + dd;
        float v = (d < 64) ? xek[d] : zsk[d - 64];
        mp += fabsf(v);
      }
      float sc = part + lane_xor(part, 16) + bzf;
      float ms = mp + lane_xor(mp, 16);
      float stored = (ms > 0.0f) ? sc : -1e38f;
      if (hh == 0) ssc[wave][r][k] = stored;
    }
    // masked softmax (both half-lanes compute redundantly)
    float mx = -1e38f;
    for (int k = 0; k < j; ++k) mx = fmaxf(mx, ssc[wave][r][k]);
    float den = 0.0f;
    for (int k = 0; k < j; ++k) {
      float sv = ssc[wave][r][k];
      float e = (sv > -1e37f) ? expf(sv - mx) : 0.0f;
      if (hh == 0) ssc[wave][r][k] = e;
      den += e;
    }
    float inv = (den > 0.0f) ? 1.0f / den : 0.0f;
    float acc[36];
#pragma unroll
    for (int dd = 0; dd < 36; ++dd) acc[dd] = 0.0f;
#pragma unroll 1
    for (int k = 0; k < j; ++k) {
      float a = ssc[wave][r][k] * inv;
      const float* xek = xe + ((size_t)srow * 32 + k) * 64;
      const float* zsk = zsbuf + ((size_t)srow * 32 + k) * 8;
#pragma unroll
      for (int dd = 0; dd < 36; ++dd) {
        int d = hh * 36 + dd;
        float v = (d < 64) ? xek[d] : zsk[d - 64];
        acc[dd] += a * v;
      }
    }
#pragma unroll
    for (int dd = 0; dd < 36; ++dd)
      arow[wave][r][128 + hh * 36 + dd] = (_Float16)acc[dd];
  }

  // MLP layer 1: (16x224) @ (224x128), ReLU, transpose to LDS
  v16h A[7];
#pragma unroll
  for (int c = 0; c < 7; ++c) A[c] = ldsA(&arow[wave][0][0], 224, lane, c * 32);
#pragma unroll 1
  for (int n = 0; n < 8; ++n) {
    float bv = b1[n * 16 + r];
    v8f acc = {bv, bv, bv, bv, bv, bv, bv, bv};
#pragma unroll
    for (int c = 0; c < 7; ++c) acc = wmma16(A[c], load_pk(pkW1, n * 7 + c, lane), acc);
#pragma unroll
    for (int v = 0; v < 8; ++v)
      hbuf[wave][v + 8 * hh][n * 16 + r] = (_Float16)fmaxf(acc[v], 0.0f);
  }
  // MLP layer 2: (16x128) @ (128x16)
  float b2v = b2[r];
  v8f acc2 = {b2v, b2v, b2v, b2v, b2v, b2v, b2v, b2v};
#pragma unroll
  for (int c = 0; c < 4; ++c) {
    v16h a2 = ldsA(&hbuf[wave][0][0], 128, lane, c * 32);
    acc2 = wmma16(a2, load_pk(pkW2, c, lane), acc2);
  }
  // outputs: cols 0..7 = mean, 8..15 = log; exchange across lane^8
  int cc = r;
#pragma unroll
  for (int v = 0; v < 8; ++v) {
    float val = acc2[v];
    float other = lane_xor(val, 8);
    int m = v + 8 * hh;
    size_t n = (size_t)(s0 + m) * 32 + j;
    if (cc < 8) {
      float mean = val, logv = other;
      float z = noise[n * 8 + cc] * expf(0.5f * logv) + mean;
      outMeans[n * 8 + cc] = mean;
      outZs[n * 8 + cc] = z;
      zswr[n * 8 + cc] = z;
    } else {
      outLogs[n * 8 + (cc - 8)] = val;
    }
  }
}

// ---------------- host launch ----------------
extern "C" void kernel_launch(void* const* d_in, const int* in_sizes, int n_in,
                              void* d_out, int out_size, void* d_ws, size_t ws_size,
                              hipStream_t stream) {
  (void)in_sizes; (void)n_in; (void)out_size; (void)ws_size;
  const float* x_y_full = (const float*)d_in[0];
  const float* x_last   = (const float*)d_in[1];
  const float* noise    = (const float*)d_in[2];
  const float* W_emb    = (const float*)d_in[3];
  const float* b_emb    = (const float*)d_in[4];
  const float* W_ih     = (const float*)d_in[5];
  const float* W_hh     = (const float*)d_in[6];
  const float* b_ih     = (const float*)d_in[7];
  const float* b_hh     = (const float*)d_in[8];
  const float* Wxy_e    = (const float*)d_in[9];
  const float* bxy_e    = (const float*)d_in[10];
  const float* Wxy_l    = (const float*)d_in[11];
  const float* bxy_l    = (const float*)d_in[12];
  const float* Wxy_c    = (const float*)d_in[13];
  const float* bxy_c    = (const float*)d_in[14];
  const float* Wxy_f    = (const float*)d_in[15];
  const float* bxy_f    = (const float*)d_in[16];
  const float* Wz_e     = (const float*)d_in[17];
  const float* bz_e     = (const float*)d_in[18];
  const float* Wz_l     = (const float*)d_in[19];
  const float* bz_l     = (const float*)d_in[20];
  const float* Wz_f     = (const float*)d_in[21];
  const float* bz_f     = (const float*)d_in[22];
  const float* W1       = (const float*)d_in[23];
  const float* b1       = (const float*)d_in[24];
  const float* W2       = (const float*)d_in[25];
  const float* b2       = (const float*)d_in[26];

  char* ws = (char*)d_ws;
  size_t off = 0;
  auto alloc = [&](size_t bytes) -> void* {
    void* p = ws + off;
    off += (bytes + 255) & ~(size_t)255;
    return p;
  };
  float* xe    = (float*)alloc((size_t)NTOT * 64 * 4);
  void*  xepk  = alloc((size_t)NTOT * 64 * 2);
  float* ae    = (float*)alloc((size_t)NTOT * 64 * 4);
  float* ac    = (float*)alloc((size_t)NTOT * 64 * 4);
  float* ze    = (float*)alloc((size_t)NTOT * 64 * 4);
  float* att   = (float*)alloc((size_t)NTOT * 64 * 4);
  float* zsbuf = (float*)alloc((size_t)SEQ * PED * 8 * 4);
  void* pkWih  = alloc((size_t)16 * 2 * 32 * 16 * 2);
  void* pkWhh  = alloc((size_t)16 * 2 * 32 * 16 * 2);
  void* pkWxye = alloc((size_t)4 * 2 * 32 * 16 * 2);
  void* pkWxyc = alloc((size_t)4 * 2 * 32 * 16 * 2);
  void* pkWze  = alloc((size_t)4 * 2 * 32 * 16 * 2);
  void* pkW1   = alloc((size_t)8 * 7 * 32 * 16 * 2);
  void* pkW2   = alloc((size_t)1 * 4 * 32 * 16 * 2);

  k_packB<<<dim3(16, 2), 32, 0, stream>>>(W_ih,  (_Float16*)pkWih,  256, 64, 2);
  k_packB<<<dim3(16, 2), 32, 0, stream>>>(W_hh,  (_Float16*)pkWhh,  256, 64, 2);
  k_packB<<<dim3(4, 2),  32, 0, stream>>>(Wxy_e, (_Float16*)pkWxye, 64, 64, 2);
  k_packB<<<dim3(4, 2),  32, 0, stream>>>(Wxy_c, (_Float16*)pkWxyc, 64, 64, 2);
  k_packB<<<dim3(4, 2),  32, 0, stream>>>(Wz_e,  (_Float16*)pkWze,  64, 64, 2);
  k_packB<<<dim3(8, 7),  32, 0, stream>>>(W1,    (_Float16*)pkW1,   128, 200, 7);
  k_packB<<<dim3(1, 4),  32, 0, stream>>>(W2,    (_Float16*)pkW2,   16, 128, 4);

  k_lstm<<<512, 256, 0, stream>>>(x_y_full, W_emb, b_emb, b_ih, b_hh,
                                  (const _Float16*)pkWih, (const _Float16*)pkWhh,
                                  xe, (_Float16*)xepk);

  k_proj<<<512, 256, 0, stream>>>((const _Float16*)xepk,
                                  (const _Float16*)pkWxye, bxy_e, ae,
                                  (const _Float16*)pkWxyc, bxy_c, ac,
                                  (const _Float16*)pkWze,  bz_e,  ze);

  k_att<<<SEQ, 256, 0, stream>>>(xe, x_last, ae, ac, Wxy_l, bxy_l, Wxy_f, bxy_f, att);

  float* out = (float*)d_out;
  float* outMeans = out;
  float* outLogs  = out + (size_t)NTOT * NDIM;
  float* outZs    = out + (size_t)2 * NTOT * NDIM;
  for (int j = 0; j < PED; ++j) {
    k_step<<<32, 128, 0, stream>>>(j, xe, x_last, ze, att, zsbuf,
                                   Wz_l, bz_l, Wz_f, bz_f,
                                   (const _Float16*)pkW1, b1,
                                   (const _Float16*)pkW2, b2,
                                   noise, outMeans, outLogs, outZs, zsbuf);
  }
}
